// GlobalEquivariantAttentionUpsample_51926154609244
// MI455X (gfx1250) — matrix operand order repaired and checked by reading, not compiled, not run on hardware
//
#include <hip/hip_runtime.h>
#include <hip/hip_bf16.h>
#include <math.h>

// ---------------------------------------------------------------------------
// GlobalEquivariantAttentionUpsample for MI455X (gfx1250, wave32, WMMA).
// B=2, C=8, N_BLOCKS=2, R_UP=4 are module constants in the reference.
// ---------------------------------------------------------------------------

typedef _Float16 h8v  __attribute__((ext_vector_type(8)));
typedef _Float16 v16h __attribute__((ext_vector_type(16)));
typedef float    v8f  __attribute__((ext_vector_type(8)));

#define NBATCH 2
#define CMUL 8
#define NBLK 2
#define RUP 4
#define DVAL 176            // C*9 + C*13
#define EPSN 1e-12f
#define KCHUNK 64           // keys staged to LDS per workgroup iteration

// 14 unique Wigner-3j tensors: 6 value paths then 8 output paths.
__constant__ int c_kL[14][3] = {
  {4,0,4},{4,2,4},{6,2,4},{6,0,6},{4,2,6},{6,2,6},
  {4,4,4},{4,6,4},{6,4,4},{6,6,4},{4,4,6},{4,6,6},{6,4,6},{6,6,6}};
__constant__ int c_kOff[14] = {0,81,486,1071,1240,1825,2670,3399,4452,5505,7026,8079,9600,11121};
#define W3_TOTAL 13318

// ------------------------------ Wigner-3j ----------------------------------
__device__ double dfact(int n) { double r = 1.0; for (int i = 2; i <= n; ++i) r *= (double)i; return r; }

__device__ double cgcoef(int j1, int m1, int j2, int m2, int j3, int m3) {
  if (m1 + m2 != m3) return 0.0;
  if (j3 < abs(j1 - j2) || j3 > j1 + j2) return 0.0;
  if (abs(m1) > j1 || abs(m2) > j2 || abs(m3) > j3) return 0.0;
  double pref = sqrt((2.0 * j3 + 1.0) * dfact(j3 + j1 - j2) * dfact(j3 - j1 + j2) *
                     dfact(j1 + j2 - j3) / dfact(j1 + j2 + j3 + 1));
  pref *= sqrt(dfact(j3 + m3) * dfact(j3 - m3) * dfact(j1 - m1) * dfact(j1 + m1) *
               dfact(j2 - m2) * dfact(j2 + m2));
  double s = 0.0;
  for (int k = 0; k <= j1 + j2 - j3; ++k) {
    int d0 = k, d1 = j1 + j2 - j3 - k, d2 = j1 - m1 - k, d3 = j2 + m2 - k;
    int d4 = j3 - j2 + m1 + k, d5 = j3 - j1 - m2 + k;
    if (d0 < 0 || d1 < 0 || d2 < 0 || d3 < 0 || d4 < 0 || d5 < 0) continue;
    double den = dfact(d0) * dfact(d1) * dfact(d2) * dfact(d3) * dfact(d4) * dfact(d5);
    s += ((k & 1) ? -1.0 : 1.0) / den;
  }
  return pref * s;
}

// Sparse rows of the real->complex SH basis change U(l): <=2 nonzeros per row.
__device__ int urow(int l, int r, int* col, double* re, double* im) {
  const double s2 = 0.7071067811865475244;
  if (r == l) { col[0] = l; re[0] = 1.0; im[0] = 0.0; return 1; }
  if (r > l) {
    int m = r - l; double sgn = (m & 1) ? -1.0 : 1.0;
    col[0] = l + m; re[0] = sgn * s2; im[0] = 0.0;
    col[1] = l - m; re[1] = s2;       im[1] = 0.0;  return 2;
  }
  int m = l - r; double sgn = (m & 1) ? -1.0 : 1.0;
  col[0] = l - m; re[0] = 0.0; im[0] = s2;
  col[1] = l + m; re[1] = 0.0; im[1] = -sgn * s2;   return 2;
}

__global__ __launch_bounds__(256) void w3j_kernel(float* __restrict__ w3) {
  const int t = blockIdx.x;
  const int l1 = c_kL[t][0], l2 = c_kL[t][1], l3 = c_kL[t][2];
  const int d1 = 2 * l1 + 1, d2 = 2 * l2 + 1, d3 = 2 * l3 + 1;
  const int total = d1 * d2 * d3;
  float* T = w3 + c_kOff[t];
  for (int e = threadIdx.x; e < total; e += blockDim.x) {
    int i = e / (d2 * d3), rem = e % (d2 * d3), j = rem / d3, k = rem % d3;
    int ca[2], cb[2], cc[2]; double ra[2], ia[2], rb[2], ib[2], rc[2], ic[2];
    int na = urow(l1, i, ca, ra, ia);
    int nb = urow(l2, j, cb, rb, ib);
    int nc = urow(l3, k, cc, rc, ic);
    double acc = 0.0;
    for (int a = 0; a < na; ++a)
      for (int b = 0; b < nb; ++b)
        for (int c = 0; c < nc; ++c) {
          double cg = cgcoef(l1, ca[a] - l1, l2, cb[b] - l2, l3, cc[c] - l3);
          if (cg == 0.0) continue;
          double pr = ra[a] * rb[b] - ia[a] * ib[b];
          double pi = ra[a] * ib[b] + ia[a] * rb[b];
          acc += cg * (pr * rc[c] + pi * ic[c]);  // times conj(U3)
        }
    T[e] = (float)acc;
  }
  __syncthreads();
  __shared__ double red[256];
  double ssq = 0.0;
  for (int e = threadIdx.x; e < total; e += blockDim.x) { double v = T[e]; ssq += v * v; }
  red[threadIdx.x] = ssq;
  __syncthreads();
  for (int s = 128; s > 0; s >>= 1) {
    if ((int)threadIdx.x < s) red[threadIdx.x] += red[threadIdx.x + s];
    __syncthreads();
  }
  float invn = (float)(1.0 / sqrt(red[0]));
  for (int e = threadIdx.x; e < total; e += blockDim.x) T[e] *= invn;
}

// ------------------------------ init feat ----------------------------------
__global__ __launch_bounds__(256) void init_feat(const float* __restrict__ f4,
                                                 const float* __restrict__ f6,
                                                 float* __restrict__ feat, int N) {
  int idx = blockIdx.x * blockDim.x + threadIdx.x;
  int tot = NBATCH * N * 22;
  if (idx >= tot) return;
  int pt = idx / 22, k = idx % 22;
  feat[idx] = (k < 9) ? f4[pt * 9 + k] : f6[pt * 13 + (k - 9)];
}

// -------------------------------- prep -------------------------------------
__device__ void val_path(const float* __restrict__ Wt, const float* fx, int dx,
                         const float* shp, int dj, int dk, float* out) {
  for (int k = 0; k < dk; ++k) out[k] = 0.f;
  for (int i = 0; i < dx; ++i) {
    float fi = fx[i];
    for (int j = 0; j < dj; ++j) {
      float fj = fi * shp[j];
      const float* row = Wt + (i * dj + j) * dk;
      for (int k = 0; k < dk; ++k) out[k] += row[k] * fj;
    }
  }
}

__global__ __launch_bounds__(256) void prep_kernel(
    const float* __restrict__ feat, const float* __restrict__ ls4,
    const float* __restrict__ ls6, const float* __restrict__ pbw,
    const float* __restrict__ pbb, const float* __restrict__ liw4,
    const float* __restrict__ liw6, const float* __restrict__ tvw,
    const float* __restrict__ w3, const int* __restrict__ Hp,
    const int* __restrict__ Wp, _Float16* __restrict__ qk,
    _Float16* __restrict__ vt, float* __restrict__ pb, int N, int blk) {
  int idx = blockIdx.x * blockDim.x + threadIdx.x;
  if (idx >= NBATCH * N) return;
  int b = idx / N, n = idx % N;
  const float* fr = feat + (size_t)idx * 22;
  float f4[9], f6[13];
  float nn4 = 0.f, nn6 = 0.f;
  for (int k = 0; k < 9; ++k)  { f4[k] = fr[k];     nn4 += f4[k] * f4[k]; }
  for (int k = 0; k < 13; ++k) { f6[k] = fr[9 + k]; nn6 += f6[k] * f6[k]; }
  float e4 = __expf(ls4[blk]), e6 = __expf(ls6[blk]);
  float r4 = sqrtf(e4) / fmaxf(sqrtf(nn4), EPSN);
  float r6 = sqrtf(e6) / fmaxf(sqrtf(nn6), EPSN);
  // packed q=k row, 22 dims padded to 32 (f16)
  _Float16* qr = qk + (size_t)idx * 32;
  for (int k = 0; k < 9; ++k)  qr[k]     = (_Float16)(f4[k] * r4);
  for (int k = 0; k < 13; ++k) qr[9 + k] = (_Float16)(f6[k] * r6);
  for (int k = 22; k < 32; ++k) qr[k] = (_Float16)0.f;

  // spherical harmonics on the LR grid (z == 0)
  int H = *Hp, W = *Wp;
  int iy = n / W, ix = n % W;
  float y = (H > 1) ? (-1.f + 2.f * (float)iy / (float)(H - 1)) : -1.f;
  float x = (W > 1) ? (-1.f + 2.f * (float)ix / (float)(W - 1)) : -1.f;
  float rr = fmaxf(sqrtf(x * x + y * y), 1e-8f);
  float xn = x / rr, yn = y / rr;
  const float c0 = 0.28209479177387814f, c2 = 1.0925484305920792f, c20 = 0.31539156525252005f;
  float sh[6];
  sh[0] = c0; sh[1] = c2 * xn * yn; sh[2] = 0.f;
  sh[3] = -c20; sh[4] = 0.f; sh[5] = 0.5f * c2 * (xn * xn - yn * yn);
  if (b == 0) {
    float pv = pbb[blk];
    for (int j = 0; j < 6; ++j) pv += sh[j] * pbw[blk * 6 + j];
    pb[n] = pv;
  }

  // value tensor products: rank-1 in channels -> g[p][w] * s[p][k]
  float sv0[9], sv1[9], sv2[9], sv3[13], sv4[13], sv5[13];
  val_path(w3 + c_kOff[0], f4, 9,  sh,     1, 9,  sv0);
  val_path(w3 + c_kOff[1], f4, 9,  sh + 1, 5, 9,  sv1);
  val_path(w3 + c_kOff[2], f6, 13, sh + 1, 5, 9,  sv2);
  val_path(w3 + c_kOff[3], f6, 13, sh,     1, 13, sv3);
  val_path(w3 + c_kOff[4], f4, 9,  sh + 1, 5, 13, sv4);
  val_path(w3 + c_kOff[5], f6, 13, sh + 1, 5, 13, sv5);
  const int xv4[6] = {1, 1, 0, 0, 1, 0};
  float g[6][8];
  for (int p = 0; p < 6; ++p) {
    const float* li = (xv4[p] ? liw4 : liw6) + blk * CMUL;
    for (int w = 0; w < 8; ++w) {
      float a = 0.f;
      for (int u = 0; u < 8; ++u) a += li[u] * tvw[(((size_t)blk * 6 + p) * 8 + u) * 8 + w];
      g[p][w] = a;
    }
  }
  const float inv_v = 0.20412414523193154f;  // 1/sqrt(3*C)
  _Float16* vb = vt + (size_t)b * DVAL * N;
  for (int w = 0; w < 8; ++w)
    for (int k = 0; k < 9; ++k)
      vb[(size_t)(w * 9 + k) * N + n] =
          (_Float16)(inv_v * (g[0][w] * sv0[k] + g[1][w] * sv1[k] + g[2][w] * sv2[k]));
  for (int w = 0; w < 8; ++w)
    for (int k = 0; k < 13; ++k)
      vb[(size_t)(72 + w * 13 + k) * N + n] =
          (_Float16)(inv_v * (g[3][w] * sv3[k] + g[4][w] * sv4[k] + g[5][w] * sv5[k]));
}

// ------------------------------ pb shift -----------------------------------
// Scores are bounded: |e4*dot4 + e6*dot6| <= e4+e6 (unit vectors). Using
// shift = e4 + e6 + max(pb) guarantees exp() never overflows -> no online max.
__global__ __launch_bounds__(256) void finalize_pb(const float* __restrict__ pb,
                                                   float* __restrict__ pbe,
                                                   const float* __restrict__ ls4,
                                                   const float* __restrict__ ls6,
                                                   int N, int blk) {
  __shared__ float red[256];
  float m = -3.0e38f;
  for (int i = threadIdx.x; i < N; i += 256) m = fmaxf(m, pb[i]);
  red[threadIdx.x] = m;
  __syncthreads();
  for (int s = 128; s > 0; s >>= 1) {
    if ((int)threadIdx.x < s) red[threadIdx.x] = fmaxf(red[threadIdx.x], red[threadIdx.x + s]);
    __syncthreads();
  }
  float shift = __expf(ls4[blk]) + __expf(ls6[blk]) + red[0];
  for (int i = threadIdx.x; i < N; i += 256) pbe[i] = pb[i] - shift;
}

// ----------------------- fused attention (WMMA + LDS) ----------------------
// Workgroup = 8 waves = 128 queries. Keys are processed in 64-key chunks that
// the whole workgroup cooperatively stages into LDS once (K rows: 4KB,
// V tiles: 22KB, bias: 256B), then every wave feeds its WMMAs from
// ds_load_b128 -- an 8x cut of the dominant L2 stream.
// Per 32-key step: 2 score WMMAs (S^T so the probability tile converts
// lane-locally into the A operand) + 11 ctx WMMAs (176 = 11 x 16 dims).
__global__ __launch_bounds__(256) void attn_kernel(
    const _Float16* __restrict__ qk,   // [B][N][32] f16 (22 dims zero-padded)
    const _Float16* __restrict__ vt,   // [B][176][N] f16 (dim-major values)
    const float* __restrict__ pbe,     // [N] shifted key bias
    float* __restrict__ ctx,           // [B][N][176] f32
    int N) {
  __shared__ _Float16 lK[KCHUNK * 32];       // 4 KB
  __shared__ _Float16 lV[DVAL * KCHUNK];     // 22 KB
  __shared__ float    lPb[KCHUNK];           // 256 B

  const int tid  = threadIdx.x;
  const int wave = tid >> 5;
  const int lane = tid & 31;
  const int l16 = lane & 15;
  const int hi = lane >> 4;
  const int tilesPerBatch = N / 128;
  const int b = blockIdx.x / tilesPerBatch;
  const int q0 = (blockIdx.x % tilesPerBatch) * 128 + wave * 16;

  const _Float16* qkb = qk + (size_t)b * N * 32;
  const _Float16* vtb = vt + (size_t)b * DVAL * N;

  // Q as B operand: lane -> query q0+l16, feature halves 8*hi & 16+8*hi
  v16h qB;
  {
    const _Float16* p = qkb + (size_t)(q0 + l16) * 32 + 8 * hi;
    *(h8v*)&qB         = *(const h8v*)p;
    *(((h8v*)&qB) + 1) = *(const h8v*)(p + 16);
  }

  const v8f vzero = {};
  v8f acc[11];
#pragma unroll
  for (int t = 0; t < 11; ++t) acc[t] = vzero;
  float rowsum = 0.f;

  for (int kc = 0; kc < N; kc += KCHUNK) {
    // ---- cooperative stage of this key chunk into LDS ----
    {
      // K rows: 64 rows x 64B = 256 x 16B segments (one per thread)
      int row = tid >> 2, seg = tid & 3;
      *(h8v*)&lK[row * 32 + seg * 8] =
          *(const h8v*)(qkb + (size_t)(kc + row) * 32 + seg * 8);
      // V tiles: 176 rows x 128B = 1408 x 16B segments
      for (int i = tid; i < DVAL * (KCHUNK / 8); i += 256) {
        int d = i >> 3, s = i & 7;
        *(h8v*)&lV[d * KCHUNK + s * 8] =
            *(const h8v*)(vtb + (size_t)d * N + kc + s * 8);
      }
      if (tid < KCHUNK) lPb[tid] = pbe[kc + tid];
    }
    __syncthreads();

    // ---- compute on the staged chunk ----
#pragma unroll
    for (int kb2 = 0; kb2 < KCHUNK; kb2 += 32) {
      // K tiles as A operand: lane -> key M=l16, feature halves 8*hi & 16+8*hi
      v16h ka0, ka1;
      {
        const _Float16* p0 = &lK[(kb2 + l16) * 32 + 8 * hi];
        *(h8v*)&ka0         = *(const h8v*)p0;
        *(((h8v*)&ka0) + 1) = *(const h8v*)(p0 + 16);
        const _Float16* p1 = &lK[(kb2 + 16 + l16) * 32 + 8 * hi];
        *(h8v*)&ka1         = *(const h8v*)p1;
        *(((h8v*)&ka1) + 1) = *(const h8v*)(p1 + 16);
      }
      // S^T tiles: D[M=key, N=query]; VGPR r -> key kc+kb2(+16) + r + 8*hi
      v8f s0 = __builtin_amdgcn_wmma_f32_16x16x32_f16(false, ka0, false, qB, (short)0, vzero, false, false);
      v8f s1 = __builtin_amdgcn_wmma_f32_16x16x32_f16(false, ka1, false, qB, (short)0, vzero, false, false);

      const float* pb0 = &lPb[kb2 + 8 * hi];
      const float* pb1 = &lPb[kb2 + 16 + 8 * hi];
      v16h pA;  // P as A operand: lane -> query M=l16; halves -> 32 keys
      float lsum = 0.f;
#pragma unroll
      for (int r = 0; r < 8; ++r) {
        float e = __expf(s0[r] + pb0[r]);
        lsum += e;
        pA[r] = (_Float16)e;
      }
#pragma unroll
      for (int r = 0; r < 8; ++r) {
        float e = __expf(s1[r] + pb1[r]);
        lsum += e;
        pA[8 + r] = (_Float16)e;
      }
      rowsum += lsum;

      // ctx accumulation: D[M=query, N=dim], V as B operand (dim-major rows)
#pragma unroll
      for (int t = 0; t < 11; ++t) {
        v16h vb;
        const _Float16* p = &lV[(16 * t + l16) * KCHUNK + kb2 + 8 * hi];
        *(h8v*)&vb         = *(const h8v*)p;
        *(((h8v*)&vb) + 1) = *(const h8v*)(p + 16);
        acc[t] = __builtin_amdgcn_wmma_f32_16x16x32_f16(false, pA, false, vb, (short)0, acc[t], false, false);
      }
    }
    __syncthreads();
  }

  // denominator per query: combine the two lane halves, then broadcast
  rowsum += __shfl_xor(rowsum, 16, 32);
  float invs = 1.f / fmaxf(rowsum, 1e-30f);
  float inv[8];
#pragma unroll
  for (int r = 0; r < 8; ++r) inv[r] = __shfl(invs, r + 8 * hi, 32);

  float* cb = ctx + ((size_t)b * N + q0) * DVAL;
#pragma unroll
  for (int t = 0; t < 11; ++t)
#pragma unroll
    for (int r = 0; r < 8; ++r)
      cb[(size_t)(r + 8 * hi) * DVAL + 16 * t + l16] = acc[t][r] * inv[r];
}

// ------------------------- output TP + residual ----------------------------
__global__ __launch_bounds__(256) void outtp_kernel(
    float* __restrict__ feat, const float* __restrict__ ctx,
    const float* __restrict__ liw4, const float* __restrict__ liw6,
    const float* __restrict__ low4, const float* __restrict__ low6,
    const float* __restrict__ tow, const float* __restrict__ w3, int N, int blk) {
  int idx = blockIdx.x * blockDim.x + threadIdx.x;
  if (idx >= NBATCH * N) return;
  float* fr = feat + (size_t)idx * 22;
  float f4[9], f6[13];
  for (int k = 0; k < 9; ++k)  f4[k] = fr[k];
  for (int k = 0; k < 13; ++k) f6[k] = fr[9 + k];
  const float* cx = ctx + (size_t)idx * DVAL;

  float d4[9]  = {0.f, 0.f, 0.f, 0.f, 0.f, 0.f, 0.f, 0.f, 0.f};
  float d6[13] = {0.f, 0.f, 0.f, 0.f, 0.f, 0.f, 0.f, 0.f, 0.f, 0.f, 0.f, 0.f, 0.f};
  const int X4[8] = {1, 1, 0, 0, 1, 1, 0, 0};
  const int Y4[8] = {1, 0, 1, 0, 1, 0, 1, 0};
  const int O4[8] = {1, 1, 1, 1, 0, 0, 0, 0};
  for (int p = 0; p < 8; ++p) {
    const float* li = (X4[p] ? liw4 : liw6) + blk * CMUL;
    const float* lo = (O4[p] ? low4 : low6) + blk * CMUL;
    // gop[v] = sum_{u,w} li[u] * tow[p][u][v][w] * lo[w]   (rank-1 h = f x li)
    float gop[8];
    for (int v = 0; v < 8; ++v) {
      float a = 0.f;
      for (int u = 0; u < 8; ++u) {
        const float* tw = tow + ((((size_t)blk * 8 + p) * 8 + u) * 8 + v) * 8;
        float lu = li[u];
        for (int w = 0; w < 8; ++w) a += lu * tw[w] * lo[w];
      }
      gop[v] = a;
    }
    int dy = Y4[p] ? 9 : 13;
    const float* cY = cx + (Y4[p] ? 0 : 72);
    float cc[13];
    for (int j = 0; j < dy; ++j) {
      float a = 0.f;
      for (int v = 0; v < 8; ++v) a += cY[v * dy + j] * gop[v];
      cc[j] = a;
    }
    const float* fx = X4[p] ? f4 : f6;
    int dxd = X4[p] ? 9 : 13;
    int dk  = O4[p] ? 9 : 13;
    float* dd = O4[p] ? d4 : d6;
    const float* Wt = w3 + c_kOff[6 + p];
    for (int i = 0; i < dxd; ++i) {
      float fi = fx[i];
      for (int j = 0; j < dy; ++j) {
        float fc = fi * cc[j];
        const float* row = Wt + (i * dy + j) * dk;
        for (int k = 0; k < dk; ++k) dd[k] += row[k] * fc;
      }
    }
  }
  const float sc = (1.f / 16.f) * 0.35355339059327373f;  // inv_o * inv_l
  for (int k = 0; k < 9; ++k)  fr[k]     = f4[k] + sc * d4[k];
  for (int k = 0; k < 13; ++k) fr[9 + k] = f6[k] + sc * d6[k];
}

// ------------------------------- upsample ----------------------------------
__global__ __launch_bounds__(256) void upsample_kernel(
    const float* __restrict__ feat, const float* __restrict__ up,
    float* __restrict__ out, const int* __restrict__ Hp,
    const int* __restrict__ Wp, int N) {
  int H = *Hp, W = *Wp;
  int HR = N * RUP * RUP;
  int idx = blockIdx.x * blockDim.x + threadIdx.x;
  if (idx >= NBATCH * HR) return;
  int b = idx / HR, p = idx % HR;
  int Wr = W * RUP;
  int Yh = p / Wr, Xh = p % Wr;
  int yl = Yh / RUP, ry = Yh % RUP, xl = Xh / RUP, rx = Xh % RUP;
  (void)H;
  const float* f = feat + ((size_t)b * N + (size_t)yl * W + xl) * 22;
  float u4 = up[(ry * RUP + rx) * 2 + 0];
  float u6 = up[(ry * RUP + rx) * 2 + 1];
  float* o4 = out + ((size_t)b * HR + p) * 9;
  float* o6 = out + (size_t)NBATCH * HR * 9 + ((size_t)b * HR + p) * 13;
  for (int k = 0; k < 9; ++k)  o4[k] = f[k] * u4;
  for (int k = 0; k < 13; ++k) o6[k] = f[9 + k] * u6;
}

// ------------------------------- launcher ----------------------------------
extern "C" void kernel_launch(void* const* d_in, const int* in_sizes, int n_in,
                              void* d_out, int out_size, void* d_ws, size_t ws_size,
                              hipStream_t stream) {
  const float* f4i  = (const float*)d_in[0];
  const float* f6i  = (const float*)d_in[1];
  const float* ls4  = (const float*)d_in[2];
  const float* ls6  = (const float*)d_in[3];
  const float* pbw  = (const float*)d_in[4];
  const float* pbb  = (const float*)d_in[5];
  const float* liw4 = (const float*)d_in[6];
  const float* liw6 = (const float*)d_in[7];
  const float* tvw  = (const float*)d_in[8];
  const float* tow  = (const float*)d_in[9];
  const float* low4 = (const float*)d_in[10];
  const float* low6 = (const float*)d_in[11];
  const float* upw  = (const float*)d_in[12];
  const int*   Hp   = (const int*)d_in[13];
  const int*   Wp   = (const int*)d_in[14];
  (void)n_in; (void)out_size; (void)ws_size;

  const int N = in_sizes[0] / (NBATCH * 9);  // 4096

  // workspace carve (256B aligned): ~10 MB total, fits easily in L2.
  char* w = (char*)d_ws;
  auto carve = [&](size_t bytes) { void* p = (void*)w; w += (bytes + 255) & ~(size_t)255; return p; };
  float*    w3   = (float*)carve((size_t)W3_TOTAL * 4);
  float*    feat = (float*)carve((size_t)NBATCH * N * 22 * 4);
  _Float16* qk   = (_Float16*)carve((size_t)NBATCH * N * 32 * 2);
  _Float16* vt   = (_Float16*)carve((size_t)NBATCH * DVAL * N * 2);
  float*    ctx  = (float*)carve((size_t)NBATCH * N * DVAL * 4);
  float*    pb   = (float*)carve((size_t)N * 4);
  float*    pbe  = (float*)carve((size_t)N * 4);

  w3j_kernel<<<14, 256, 0, stream>>>(w3);
  init_feat<<<(NBATCH * N * 22 + 255) / 256, 256, 0, stream>>>(f4i, f6i, feat, N);

  for (int blk = 0; blk < NBLK; ++blk) {
    prep_kernel<<<(NBATCH * N + 255) / 256, 256, 0, stream>>>(
        feat, ls4, ls6, pbw, pbb, liw4, liw6, tvw, w3, Hp, Wp, qk, vt, pb, N, blk);
    finalize_pb<<<1, 256, 0, stream>>>(pb, pbe, ls4, ls6, N, blk);
    attn_kernel<<<NBATCH * (N / 128), 256, 0, stream>>>(qk, vt, pbe, ctx, N);
    outtp_kernel<<<(NBATCH * N + 255) / 256, 256, 0, stream>>>(
        feat, ctx, liw4, liw6, low4, low6, tow, w3, N, blk);
  }
  upsample_kernel<<<(NBATCH * N * RUP * RUP + 255) / 256, 256, 0, stream>>>(
      feat, upw, (float*)d_out, Hp, Wp, N);
}
